// MbpGINELayer_24824910970958
// MI455X (gfx1250) — matrix operand order repaired and checked by reading, not compiled, not run on hardware
//
#include <hip/hip_runtime.h>

// MbpGINELayer for MI455X (gfx1250): fused WMMA f16 GEMMs + L2-resident
// gather/scatter. N=50000 nodes, E=640000 edges, H=128.
//
// Workspace: Qh[N*H] f32 | Kh[N*H] f32 | agg[N*H] f32 | 5x f16 weights (padded)
// Output: [N*H] layernormed h, then [E*H] layernormed e.

#define N_NODES 50000
#define N_EDGES 640000
#define HD      128
#define WS_STRIDE 136   // halves per row (272B = 17*16B -> 16B-aligned rows, bank-skewed)
#define AS_STRIDE 136
#define CS_STRIDE 132   // floats per row (528B, 16B-aligned, bank-skewed)

typedef __attribute__((ext_vector_type(16))) _Float16 v16h;
typedef __attribute__((ext_vector_type(8)))  _Float16 h8v;
typedef __attribute__((ext_vector_type(8)))  float    v8f;
typedef __attribute__((ext_vector_type(4)))  float    f4v;
typedef __attribute__((ext_vector_type(4)))  int      i4v;

#if __has_builtin(__builtin_amdgcn_global_load_async_to_lds_b128)
#define HAVE_ASYNC_LDS 1
#else
#define HAVE_ASYNC_LDS 0
#endif

// 16B global -> LDS copy. Async (ASYNCcnt-tracked, no VGPR round trip) when the
// toolchain exposes the gfx1250 builtin; plain vector copy otherwise.
__device__ __forceinline__ void copy16_g2l(const _Float16* g, _Float16* l) {
#if HAVE_ASYNC_LDS
    __builtin_amdgcn_global_load_async_to_lds_b128((i4v*)g, (i4v*)l, 0, 0);
#else
    *(h8v*)l = *(const h8v*)g;
#endif
}

__device__ __forceinline__ void async_fence() {
#if HAVE_ASYNC_LDS
#if __has_builtin(__builtin_amdgcn_s_wait_asynccnt)
    __builtin_amdgcn_s_wait_asynccnt(0);
#else
    asm volatile("s_wait_asynccnt 0x0" ::: "memory");
#endif
#endif
}

// Stage one pre-converted f16 weight matrix (HD x WS_STRIDE halves, identical
// layout in global and LDS) into LDS.
__device__ __forceinline__ void stage_weight(const _Float16* __restrict__ g,
                                             _Float16* l, int tid) {
    for (int idx = tid; idx < HD * WS_STRIDE / 8; idx += 256)
        copy16_g2l(g + idx * 8, l + idx * 8);
}

// A-fragment (16x32 f16, MxK): lane L holds row M=L%16; lanes 0-15 carry
// K = {k0..k0+7, k0+16..k0+23}, lanes 16-31 carry K = {k0+8.., k0+24..}.
__device__ __forceinline__ v16h load_a_frag(const _Float16* As, int rowBase, int k0, int lane) {
    int m  = rowBase + (lane & 15);
    int kh = (lane & 16) ? 8 : 0;
    const h8v* p = (const h8v*)(As + m * AS_STRIDE + k0 + kh);   // 16B aligned
    h8v lo = p[0];
    h8v hi = p[2];
    return __builtin_shufflevector(lo, hi, 0,1,2,3,4,5,6,7,8,9,10,11,12,13,14,15);
}

// B-fragment (32x16 f16, KxN) for C = A @ W^T: B[k][n] = W[n][k].
__device__ __forceinline__ v16h load_b_frag(const _Float16* Ws, int colBase, int k0, int lane) {
    int n  = colBase + (lane & 15);
    int kk = k0 + ((lane & 16) ? 8 : 0);
    const h8v* p = (const h8v*)(Ws + n * WS_STRIDE + kk);        // 16B aligned
    return __builtin_shufflevector(p[0], p[2], 0,1,2,3,4,5,6,7,8,9,10,11,12,13,14,15);
}

// One wave computes a 64x16 strip of C = A(64x128) @ W^T(128x128).
__device__ __forceinline__ void wmma_gemm_strip(const _Float16* As, const _Float16* Ws,
                                                int colBase, int lane, v8f acc[4]) {
#pragma unroll
    for (int ks = 0; ks < 4; ++ks) {
        v16h b = load_b_frag(Ws, colBase, ks * 32, lane);
#pragma unroll
        for (int rt = 0; rt < 4; ++rt) {
            v16h a = load_a_frag(As, rt * 16, ks * 32, lane);
            acc[rt] = __builtin_amdgcn_wmma_f32_16x16x32_f16(
                false, a, false, b, (short)0, acc[rt], false, false);
        }
    }
}

__global__ void zero_kernel(float* __restrict__ p, int n4) {
    int i = blockIdx.x * blockDim.x + threadIdx.x;
    if (i < n4) ((f4v*)p)[i] = (f4v){0.f, 0.f, 0.f, 0.f};
}

// f32 HxH weight -> f16 with WS_STRIDE row padding (LDS-image layout).
__global__ void cvt_w_kernel(const float* __restrict__ W, _Float16* __restrict__ out) {
    int idx = blockIdx.x * 256 + threadIdx.x;           // < HD*HD
    int r = idx >> 7, c = idx & 127;
    out[r * WS_STRIDE + c] = (_Float16)W[idx];
}

// Qh = x@Wq^T + bq ; Kh = x@Wk^T + bk   (x tile staged once, both weights in LDS)
__global__ void qk_kernel(const float* __restrict__ x,
                          const _Float16* __restrict__ WqH, const float* __restrict__ bq,
                          const _Float16* __restrict__ WkH, const float* __restrict__ bk,
                          float* __restrict__ Qh, float* __restrict__ Kh) {
    extern __shared__ char smem[];
    _Float16* WsQ = (_Float16*)smem;
    _Float16* WsK = WsQ + HD * WS_STRIDE;
    _Float16* As  = WsK + HD * WS_STRIDE;
    int tid = threadIdx.x, lane = tid & 31, wave = tid >> 5;
    int rowBlk = blockIdx.x * 64;

    stage_weight(WqH, WsQ, tid);
    stage_weight(WkH, WsK, tid);
    for (int idx = tid; idx < 64 * (HD / 4); idx += 256) {
        int r = idx >> 5, c4 = (idx & 31) * 4;
        int g = rowBlk + r;
        f4v v = (g < N_NODES) ? *(const f4v*)&x[(size_t)g * HD + c4]
                              : (f4v){0.f, 0.f, 0.f, 0.f};
        _Float16* d = &As[r * AS_STRIDE + c4];
        d[0] = (_Float16)v[0]; d[1] = (_Float16)v[1];
        d[2] = (_Float16)v[2]; d[3] = (_Float16)v[3];
    }
    async_fence();
    __syncthreads();

    int colBase = wave << 4;
    int col  = colBase + (lane & 15);
    int rOff = (lane & 16) ? 8 : 0;
    v8f accq[4] = {}; v8f acck[4] = {};
    wmma_gemm_strip(As, WsQ, colBase, lane, accq);
    wmma_gemm_strip(As, WsK, colBase, lane, acck);

    float bqv = bq[col], bkv = bk[col];
#pragma unroll
    for (int rt = 0; rt < 4; ++rt)
#pragma unroll
        for (int r = 0; r < 8; ++r) {
            int g = rowBlk + rt * 16 + r + rOff;
            if (g < N_NODES) {
                Qh[(size_t)g * HD + col] = accq[rt][r] + bqv;
                Kh[(size_t)g * HD + col] = acck[rt][r] + bkv;
            }
        }
}

// Per 64-edge tile: Eh = conn@We^T ; relu(Eh + Qh[dst] + Kh[src]) @ Wc^T + bc ;
// atomic scatter into agg ; e = LN(conn + conn_out) -> out_e.
__global__ void edge_kernel(const float* __restrict__ conn,
                            const int*   __restrict__ edge_index,
                            const _Float16* __restrict__ WeH, const _Float16* __restrict__ WcH,
                            const float* __restrict__ bc,
                            const float* __restrict__ Qh, const float* __restrict__ Kh,
                            const float* __restrict__ g_e, const float* __restrict__ b_e,
                            float* __restrict__ agg, float* __restrict__ out_e) {
    extern __shared__ char smem[];
    _Float16* WsE = (_Float16*)smem;
    _Float16* WsC = WsE + HD * WS_STRIDE;
    _Float16* As  = WsC + HD * WS_STRIDE;
    float*    Cs  = (float*)(As + 64 * AS_STRIDE);
    int*      dstl = (int*)(Cs + 64 * CS_STRIDE);
    int*      srcl = dstl + 64;
    int tid = threadIdx.x, lane = tid & 31, wave = tid >> 5;
    int eBase = blockIdx.x * 64;

    stage_weight(WeH, WsE, tid);
    stage_weight(WcH, WsC, tid);
    for (int idx = tid; idx < 64 * (HD / 4); idx += 256) {
        int r = idx >> 5, c4 = (idx & 31) * 4;
        f4v v = *(const f4v*)&conn[(size_t)(eBase + r) * HD + c4];
        _Float16* d = &As[r * AS_STRIDE + c4];
        d[0] = (_Float16)v[0]; d[1] = (_Float16)v[1];
        d[2] = (_Float16)v[2]; d[3] = (_Float16)v[3];
    }
    if (tid < 64) {
        dstl[tid] = edge_index[eBase + tid];            // row 0 of edge_index
        srcl[tid] = edge_index[N_EDGES + eBase + tid];  // row 1
    }
    async_fence();
    __syncthreads();

    int colBase = wave << 4;
    int col  = colBase + (lane & 15);
    int rOff = (lane & 16) ? 8 : 0;

    v8f acc[4] = {};
    wmma_gemm_strip(As, WsE, colBase, lane, acc);       // Eh fragments
    __syncthreads();                                    // all As reads done

    // conn1 = Eh + Qh[dst] + Kh[src]; relu; stage conn2 back into As (f16).
    // Qh/Kh rows are random but L2-resident (51 MB); half-wave lanes share a row
    // so the 16 scalar loads coalesce into 64B segments.
#pragma unroll
    for (int rt = 0; rt < 4; ++rt)
#pragma unroll
        for (int r = 0; r < 8; ++r) {
            int lrow = rt * 16 + r + rOff;
            float v = acc[rt][r] + Qh[(size_t)dstl[lrow] * HD + col]
                                 + Kh[(size_t)srcl[lrow] * HD + col];
            As[lrow * AS_STRIDE + col] = (_Float16)fmaxf(v, 0.f);
        }
    __syncthreads();

    v8f acc2[4] = {};
    wmma_gemm_strip(As, WsC, colBase, lane, acc2);      // conn_out fragments
    float bcv = bc[col];
#pragma unroll
    for (int rt = 0; rt < 4; ++rt)
#pragma unroll
        for (int r = 0; r < 8; ++r) {
            int lrow = rt * 16 + r + rOff;
            Cs[lrow * CS_STRIDE + col] = acc2[rt][r] + bcv;
        }
    __syncthreads();

    // Scatter-add into agg (hw f32 atomics, L2-resident target) + edge residual.
    for (int idx = tid; idx < 64 * (HD / 4); idx += 256) {
        int r = idx >> 5, c4 = (idx & 31) * 4;
        float* cp = &Cs[r * CS_STRIDE + c4];
        f4v v = *(const f4v*)cp;
        float* ap = &agg[(size_t)dstl[r] * HD + c4];
        unsafeAtomicAdd(ap + 0, v[0]);
        unsafeAtomicAdd(ap + 1, v[1]);
        unsafeAtomicAdd(ap + 2, v[2]);
        unsafeAtomicAdd(ap + 3, v[3]);
        f4v cn = *(const f4v*)&conn[(size_t)(eBase + r) * HD + c4];
        *(f4v*)cp = (f4v){v[0] + cn[0], v[1] + cn[1], v[2] + cn[2], v[3] + cn[3]};
    }
    __syncthreads();

    // LayerNorm: 4 lanes per row, shuffle reduction within the quad.
    int row = tid >> 2, sub = tid & 3;
    const float* crow = &Cs[row * CS_STRIDE + sub * 32];
    float s = 0.f, ss = 0.f;
#pragma unroll
    for (int i = 0; i < 8; ++i) {
        f4v v = *(const f4v*)&crow[i * 4];
        s  += v[0] + v[1] + v[2] + v[3];
        ss += v[0] * v[0] + v[1] * v[1] + v[2] * v[2] + v[3] * v[3];
    }
    s  += __shfl_xor(s, 1);  s  += __shfl_xor(s, 2);
    ss += __shfl_xor(ss, 1); ss += __shfl_xor(ss, 2);
    float mean = s * (1.f / 128.f);
    float rstd = rsqrtf(ss * (1.f / 128.f) - mean * mean + 1e-5f);
#pragma unroll
    for (int i = 0; i < 8; ++i) {
        int c = sub * 32 + i * 4;
        f4v v = *(const f4v*)&Cs[row * CS_STRIDE + c];
        f4v gg = *(const f4v*)&g_e[c];
        f4v bb = *(const f4v*)&b_e[c];
        f4v o;
        o[0] = (v[0] - mean) * rstd * gg[0] + bb[0];
        o[1] = (v[1] - mean) * rstd * gg[1] + bb[1];
        o[2] = (v[2] - mean) * rstd * gg[2] + bb[2];
        o[3] = (v[3] - mean) * rstd * gg[3] + bb[3];
        *(f4v*)&out_e[(size_t)(eBase + row) * HD + c] = o;
    }
}

// h = LN( x + (agg*dc0 + agg*sqrt_deg*dc1) @ Wn^T + bn )
__global__ void node_kernel(const float* __restrict__ x, const float* __restrict__ agg,
                            const float* __restrict__ sqrt_deg,
                            const _Float16* __restrict__ WnH, const float* __restrict__ bn,
                            const float* __restrict__ deg_coef,
                            const float* __restrict__ g_h, const float* __restrict__ b_h,
                            float* __restrict__ out_h) {
    extern __shared__ char smem[];
    _Float16* Ws = (_Float16*)smem;
    _Float16* As = Ws + HD * WS_STRIDE;
    float*    Cs = (float*)(As + 64 * AS_STRIDE);
    int tid = threadIdx.x, lane = tid & 31, wave = tid >> 5;
    int rowBlk = blockIdx.x * 64;

    stage_weight(WnH, Ws, tid);
    for (int idx = tid; idx < 64 * (HD / 4); idx += 256) {
        int r = idx >> 5, c4 = (idx & 31) * 4;
        int g = rowBlk + r;
        _Float16* d = &As[r * AS_STRIDE + c4];
        if (g < N_NODES) {
            f4v a   = *(const f4v*)&agg[(size_t)g * HD + c4];
            f4v d01 = *(const f4v*)&deg_coef[2 * c4];
            f4v d23 = *(const f4v*)&deg_coef[2 * c4 + 4];
            float sd = sqrt_deg[g];
            d[0] = (_Float16)(a[0] * d01[0] + a[0] * sd * d01[1]);
            d[1] = (_Float16)(a[1] * d01[2] + a[1] * sd * d01[3]);
            d[2] = (_Float16)(a[2] * d23[0] + a[2] * sd * d23[1]);
            d[3] = (_Float16)(a[3] * d23[2] + a[3] * sd * d23[3]);
        } else {
            d[0] = d[1] = d[2] = d[3] = (_Float16)0.f;
        }
    }
    async_fence();
    __syncthreads();

    int colBase = wave << 4;
    int col  = colBase + (lane & 15);
    int rOff = (lane & 16) ? 8 : 0;
    v8f acc[4] = {};
    wmma_gemm_strip(As, Ws, colBase, lane, acc);

    float bnv = bn[col];
#pragma unroll
    for (int rt = 0; rt < 4; ++rt)
#pragma unroll
        for (int r = 0; r < 8; ++r) {
            int lrow = rt * 16 + r + rOff;
            int g = rowBlk + lrow;
            float xr = (g < N_NODES) ? x[(size_t)g * HD + col] : 0.f;
            Cs[lrow * CS_STRIDE + col] = acc[rt][r] + bnv + xr;
        }
    __syncthreads();

    int row = tid >> 2, sub = tid & 3;
    int g = rowBlk + row;
    const float* crow = &Cs[row * CS_STRIDE + sub * 32];
    float s = 0.f, ss = 0.f;
#pragma unroll
    for (int i = 0; i < 8; ++i) {
        f4v v = *(const f4v*)&crow[i * 4];
        s  += v[0] + v[1] + v[2] + v[3];
        ss += v[0] * v[0] + v[1] * v[1] + v[2] * v[2] + v[3] * v[3];
    }
    s  += __shfl_xor(s, 1);  s  += __shfl_xor(s, 2);
    ss += __shfl_xor(ss, 1); ss += __shfl_xor(ss, 2);
    float mean = s * (1.f / 128.f);
    float rstd = rsqrtf(ss * (1.f / 128.f) - mean * mean + 1e-5f);
    if (g < N_NODES) {
#pragma unroll
        for (int i = 0; i < 8; ++i) {
            int c = sub * 32 + i * 4;
            f4v v = *(const f4v*)&Cs[row * CS_STRIDE + c];
            f4v gg = *(const f4v*)&g_h[c];
            f4v bb = *(const f4v*)&b_h[c];
            f4v o;
            o[0] = (v[0] - mean) * rstd * gg[0] + bb[0];
            o[1] = (v[1] - mean) * rstd * gg[1] + bb[1];
            o[2] = (v[2] - mean) * rstd * gg[2] + bb[2];
            o[3] = (v[3] - mean) * rstd * gg[3] + bb[3];
            *(f4v*)&out_h[(size_t)g * HD + c] = o;
        }
    }
}

extern "C" void kernel_launch(void* const* d_in, const int* in_sizes, int n_in,
                              void* d_out, int out_size, void* d_ws, size_t ws_size,
                              hipStream_t stream) {
    const float* x          = (const float*)d_in[0];
    const float* conn       = (const float*)d_in[1];
    const int*   edge_index = (const int*)  d_in[2];
    const float* sqrt_deg   = (const float*)d_in[3];
    const float* Wq = (const float*)d_in[4];
    const float* bq = (const float*)d_in[5];
    const float* Wk = (const float*)d_in[6];
    const float* bk = (const float*)d_in[7];
    const float* We = (const float*)d_in[8];
    const float* Wc = (const float*)d_in[9];
    const float* bc = (const float*)d_in[10];
    const float* Wn = (const float*)d_in[11];
    const float* bn = (const float*)d_in[12];
    const float* deg_coef = (const float*)d_in[13];
    const float* g_h = (const float*)d_in[14];
    const float* b_h = (const float*)d_in[15];
    const float* g_e = (const float*)d_in[16];
    const float* b_e = (const float*)d_in[17];

    float* out_h = (float*)d_out;
    float* out_e = out_h + (size_t)N_NODES * HD;

    float* Qh  = (float*)d_ws;                    // N*H f32
    float* Kh  = Qh + (size_t)N_NODES * HD;
    float* agg = Kh + (size_t)N_NODES * HD;
    _Float16* Wf = (_Float16*)(agg + (size_t)N_NODES * HD);   // 5 padded f16 mats
    _Float16* WqH = Wf + 0 * HD * WS_STRIDE;
    _Float16* WkH = Wf + 1 * HD * WS_STRIDE;
    _Float16* WeH = Wf + 2 * HD * WS_STRIDE;
    _Float16* WcH = Wf + 3 * HD * WS_STRIDE;
    _Float16* WnH = Wf + 4 * HD * WS_STRIDE;

    constexpr int SMEM_QK   = 2 * HD * WS_STRIDE * 2 + 64 * AS_STRIDE * 2;                      // 87040
    constexpr int SMEM_EDGE = 2 * HD * WS_STRIDE * 2 + 64 * AS_STRIDE * 2
                              + 64 * CS_STRIDE * 4 + 2 * 64 * 4;                                // 121344
    constexpr int SMEM_NODE = HD * WS_STRIDE * 2 + 64 * AS_STRIDE * 2 + 64 * CS_STRIDE * 4;     // 86016

    const int nBlocksN = (N_NODES + 63) / 64;     // 782
    const int nBlocksE = N_EDGES / 64;            // 10000
    const int n4 = N_NODES * HD / 4;
    const int wBlocks = HD * HD / 256;            // 64

    zero_kernel <<<(n4 + 255) / 256, 256, 0, stream>>>(agg, n4);
    cvt_w_kernel<<<wBlocks, 256, 0, stream>>>(Wq, WqH);
    cvt_w_kernel<<<wBlocks, 256, 0, stream>>>(Wk, WkH);
    cvt_w_kernel<<<wBlocks, 256, 0, stream>>>(We, WeH);
    cvt_w_kernel<<<wBlocks, 256, 0, stream>>>(Wc, WcH);
    cvt_w_kernel<<<wBlocks, 256, 0, stream>>>(Wn, WnH);
    qk_kernel  <<<nBlocksN, 256, SMEM_QK,   stream>>>(x, WqH, bq, WkH, bk, Qh, Kh);
    edge_kernel<<<nBlocksE, 256, SMEM_EDGE, stream>>>(conn, edge_index, WeH, WcH, bc,
                                                      Qh, Kh, g_e, b_e, agg, out_e);
    node_kernel<<<nBlocksN, 256, SMEM_NODE, stream>>>(x, agg, sqrt_deg, WnH, bn,
                                                      deg_coef, g_h, b_h, out_h);
}
